// Block_12326556140026
// MI455X (gfx1250) — matrix operand order repaired and checked by reading, not compiled, not run on hardware
//
#include <hip/hip_runtime.h>

typedef __bf16 bf16_t;
typedef __attribute__((ext_vector_type(8)))  bf16_t v8bf;
typedef __attribute__((ext_vector_type(16))) bf16_t v16bf;
typedef __attribute__((ext_vector_type(8)))  float  v8f;
typedef __attribute__((ext_vector_type(4)))  int    v4i;

#define DIM 768
#define HEADS 12
#define DH 64
#define HIDDEN 3072
#define SEQ 1024
#define BATCH 8
#define ROWS (BATCH * SEQ) /* 8192 */

__device__ inline v8f wmma_bf16(v16bf a, v16bf b, v8f c) {
  // D = A(16x32) * B(32x16) + C, f32 accum
  return __builtin_amdgcn_wmma_f32_16x16x32_bf16(false, a, false, b, (short)0, c,
                                                 false, false);
}

// ---------------------------------------------------------------------------
// Async global -> LDS copy (16 bytes / lane), ASYNCcnt-tracked (CDNA5).
// Builtin prototype (from hipcc diagnostic): arg1 = int4 in AS(1) (global),
// arg2 = int4 in AS(3) (LDS), then offset + cpol.
// ---------------------------------------------------------------------------
__device__ inline void async_copy128(bf16_t* lds, const bf16_t* gsrc) {
#if __has_builtin(__builtin_amdgcn_global_load_async_to_lds_b128)
  __builtin_amdgcn_global_load_async_to_lds_b128(
      (__attribute__((address_space(1))) v4i*)gsrc,
      (__attribute__((address_space(3))) v4i*)lds, 0, 0);
#else
  asm volatile("global_load_async_to_lds_b128 %0, %1, off"
               :
               : "v"((unsigned)(unsigned long long)(
                     __attribute__((address_space(3))) void*)lds),
                 "v"((unsigned long long)gsrc)
               : "memory");
#endif
}

__device__ inline void wait_async0() {
#if __has_builtin(__builtin_amdgcn_s_wait_asynccnt)
  __builtin_amdgcn_s_wait_asynccnt(0);
#else
  asm volatile("s_wait_asynccnt 0x0" ::: "memory");
#endif
}

// A-fragment (16x32 bf16). Lane L holds row L%16.
// halves 0..7  -> K = (L/16)*8 + h
// halves 8..15 -> K = 16 + (L/16)*8 + (h-8)
__device__ inline v16bf frag_a(const bf16_t* rowp, int hi) {
  v8bf a = *(const v8bf*)(rowp + hi * 8);
  v8bf b = *(const v8bf*)(rowp + 16 + hi * 8);
  v16bf r;
#pragma unroll
  for (int i = 0; i < 8; ++i) { r[i] = a[i]; r[i + 8] = b[i]; }
  return r;
}

// B-fragment (32x16 bf16). Lane L holds column L%16.
// halves 0..7  -> K = (L/16)*8 + h
// halves 8..15 -> K = 16 + (L/16)*8 + (h-8)
__device__ inline v16bf frag_b(const bf16_t* colp, int hi) {
  v8bf a = *(const v8bf*)(colp + hi * 8);
  v8bf b = *(const v8bf*)(colp + 16 + hi * 8);
  v16bf r;
#pragma unroll
  for (int i = 0; i < 8; ++i) { r[i] = a[i]; r[i + 8] = b[i]; }
  return r;
}

// ---------------------------------------------------------------------------
// Weight convert + transpose: W[K][N] fp32 -> Wt[N][K] bf16
// ---------------------------------------------------------------------------
__global__ void convert_wt(const float* __restrict__ W, bf16_t* __restrict__ Wt,
                           int K, int N) {
  const size_t i = (size_t)blockIdx.x * 256 + threadIdx.x;
  if (i >= (size_t)K * N) return;
  const int n = (int)(i / K);
  const int kk = (int)(i % K);
  Wt[i] = (bf16_t)W[(size_t)kk * N + n];
}

// ---------------------------------------------------------------------------
// LayerNorm, one wave per row (DIM=768 -> 24 elems/lane), bf16 output
// ---------------------------------------------------------------------------
__global__ __launch_bounds__(256) void ln_kernel(const float* __restrict__ x,
                                                 const float* __restrict__ w,
                                                 const float* __restrict__ b,
                                                 bf16_t* __restrict__ out) {
  const int lane = threadIdx.x & 31;
  const int wid = threadIdx.x >> 5;
  const size_t row = (size_t)blockIdx.x * 8 + wid;
  const float* xr = x + row * DIM;
  float vals[24];
  float s = 0.f;
#pragma unroll
  for (int i = 0; i < 24; ++i) { vals[i] = xr[i * 32 + lane]; s += vals[i]; }
  for (int m = 16; m >= 1; m >>= 1) s += __shfl_xor(s, m, 32);
  const float mu = s * (1.0f / DIM);
  float vs = 0.f;
#pragma unroll
  for (int i = 0; i < 24; ++i) { const float d = vals[i] - mu; vs += d * d; }
  for (int m = 16; m >= 1; m >>= 1) vs += __shfl_xor(vs, m, 32);
  const float inv = rsqrtf(vs * (1.0f / DIM) + 1e-5f);
  bf16_t* orow = out + row * DIM;
#pragma unroll
  for (int i = 0; i < 24; ++i) {
    const int c = i * 32 + lane;
    orow[c] = (bf16_t)((vals[i] - mu) * inv * w[c] + b[c]);
  }
}

// ---------------------------------------------------------------------------
// Generic bf16 WMMA GEMM: C[M,N] = A[M,K] * Bt[N,K]^T (+ bias, epilogue)
// Block = 256 threads = 8 waves; tile 128x128; wave tile 64x32; K-step 32.
// Double-buffered LDS with async global->LDS prefetch of the next K-slab.
// EPI 0: scatter to q (*0.125) / k / v  [B,H,N,64] bf16
// EPI 1: of = resid + (acc+bias)*gamma  (fp32, [M,N])
// EPI 2: o0 = gelu(acc+bias)            (bf16, [M,N])
// ---------------------------------------------------------------------------
template <int EPI>
__global__ __launch_bounds__(256) void gemm_bf16(
    const bf16_t* __restrict__ A, const bf16_t* __restrict__ Bt,
    const float* __restrict__ bias, int M, int N, int K,
    bf16_t* __restrict__ o0, bf16_t* __restrict__ o1, bf16_t* __restrict__ o2,
    float* __restrict__ of, const float* __restrict__ resid,
    const float* __restrict__ gamma) {
  __shared__ bf16_t As[2][128 * 32];
  __shared__ bf16_t Bs[2][128 * 32];
  const int tid = threadIdx.x;
  const int lane = tid & 31;
  const int lo = lane & 15, hi = lane >> 4;
  const int wid = tid >> 5;
  const int wm = (wid >> 2) * 64;
  const int wn = (wid & 3) * 32;
  const int bm = blockIdx.x * 128;
  const int bn = blockIdx.y * 128;

  v8f acc[4][2] = {};

  const int lr = tid >> 1;        // 0..127: tile row this thread stages
  const int lc = (tid & 1) * 16;  // 0 or 16: element offset within k-slab
  const bf16_t* gA = A + (size_t)(bm + lr) * K + lc;
  const bf16_t* gB = Bt + (size_t)(bn + lr) * K + lc;
  const int ldst = lr * 32 + lc;

  // prologue: stage K-slab 0 into buffer 0
  async_copy128(&As[0][ldst], gA);
  async_copy128(&As[0][ldst + 8], gA + 8);
  async_copy128(&Bs[0][ldst], gB);
  async_copy128(&Bs[0][ldst + 8], gB + 8);

  const int nk = K >> 5;
  for (int it = 0; it < nk; ++it) {
    wait_async0();
    __syncthreads();  // current slab resident, all waves arrived
    const int buf = it & 1;
    if (it + 1 < nk) {  // prefetch next slab into the other buffer
      const bf16_t* ga = gA + (it + 1) * 32;
      const bf16_t* gb = gB + (it + 1) * 32;
      async_copy128(&As[buf ^ 1][ldst], ga);
      async_copy128(&As[buf ^ 1][ldst + 8], ga + 8);
      async_copy128(&Bs[buf ^ 1][ldst], gb);
      async_copy128(&Bs[buf ^ 1][ldst + 8], gb + 8);
    }

    const bf16_t* as = As[buf];
    const bf16_t* bs = Bs[buf];
    v16bf af[4], bfr[2];
#pragma unroll
    for (int i = 0; i < 4; ++i) af[i] = frag_a(as + (wm + i * 16 + lo) * 32, hi);
#pragma unroll
    for (int j = 0; j < 2; ++j) bfr[j] = frag_b(bs + (wn + j * 16 + lo) * 32, hi);
#pragma unroll
    for (int i = 0; i < 4; ++i)
#pragma unroll
      for (int j = 0; j < 2; ++j) acc[i][j] = wmma_bf16(af[i], bfr[j], acc[i][j]);
    __syncthreads();  // everyone done reading buf before it is re-filled
  }

#pragma unroll
  for (int i = 0; i < 4; ++i) {
#pragma unroll
    for (int j = 0; j < 2; ++j) {
      const int col = bn + wn + j * 16 + lo;
      const float bv = bias[col];
#pragma unroll
      for (int jj = 0; jj < 8; ++jj) {
        const int row = bm + wm + i * 16 + jj + 8 * hi;
        const float v = acc[i][j][jj] + bv;
        if (EPI == 0) {
          const int which = col / DIM;
          const int w2 = col - which * DIM;
          const int head = w2 >> 6;
          const int d = w2 & 63;
          const int b_ = row >> 10;
          const int n_ = row & 1023;
          const size_t idx = (((size_t)(b_ * HEADS + head)) * SEQ + n_) * DH + d;
          if (which == 0)      o0[idx] = (bf16_t)(v * 0.125f);  // q * Dh^-0.5
          else if (which == 1) o1[idx] = (bf16_t)v;             // k
          else                 o2[idx] = (bf16_t)v;             // v
        } else if (EPI == 1) {
          const size_t idx = (size_t)row * N + col;
          of[idx] = resid[idx] + v * gamma[col];
        } else {
          const float g = 0.5f * v * (1.0f + erff(v * 0.70710678118f));
          o0[(size_t)row * N + col] = (bf16_t)g;
        }
      }
    }
  }
}

// ---------------------------------------------------------------------------
// Flash attention: grid (B*H, SEQ/128). 8 waves/block, 16 query rows/wave.
// Key tiles of 32 double-buffered in LDS via async copies; online softmax;
// P restaged via LDS to get the A-fragment layout for the P*V WMMA.
// ---------------------------------------------------------------------------
__global__ __launch_bounds__(256) void attn_kernel(const bf16_t* __restrict__ q,
                                                   const bf16_t* __restrict__ k,
                                                   const bf16_t* __restrict__ v,
                                                   bf16_t* __restrict__ o) {
  __shared__ bf16_t Kt[2][32 * 64];
  __shared__ bf16_t Vt[2][32 * 64];
  __shared__ bf16_t Pl[8][16 * 32];

  const int tid = threadIdx.x;
  const int lane = tid & 31, lo = lane & 15, hi = lane >> 4;
  const int wid = tid >> 5;
  const int bh = blockIdx.x;  // 0..95  (b*HEADS + h)
  const int rt = blockIdx.y;  // 0..7
  const size_t base = (size_t)bh * SEQ * DH;
  const int row0 = rt * 128 + wid * 16;

  const bf16_t* qrow = q + base + (size_t)(row0 + lo) * DH;
  const v16bf qf0 = frag_a(qrow, hi);
  const v16bf qf1 = frag_a(qrow + 32, hi);

  float mrow[8], lrow[8];
#pragma unroll
  for (int j = 0; j < 8; ++j) { mrow[j] = -1e30f; lrow[j] = 0.f; }
  v8f oacc[4] = {};

  // prologue: key tile 0 into buffer 0 (one b128/lane per tile)
  async_copy128(&Kt[0][tid * 8], k + base + tid * 8);
  async_copy128(&Vt[0][tid * 8], v + base + tid * 8);

  for (int it = 0; it < SEQ / 32; ++it) {
    wait_async0();
    __syncthreads();  // tile resident; prior tile's readers all finished
    const int buf = it & 1;
    if (it + 1 < SEQ / 32) {
      const size_t goff = base + (size_t)(it + 1) * 32 * DH + tid * 8;
      async_copy128(&Kt[buf ^ 1][tid * 8], k + goff);
      async_copy128(&Vt[buf ^ 1][tid * 8], v + goff);
    }

    // S(16x32) = Q(16x64) * K_tile^T(64x32), two 16-col chunks
    v8f s[2];
#pragma unroll
    for (int nc = 0; nc < 2; ++nc) {
      const bf16_t* kr = Kt[buf] + (nc * 16 + lo) * 64;  // row of K = col of K^T
      v8f z = {};
      z = wmma_bf16(qf0, frag_b(kr, hi), z);
      z = wmma_bf16(qf1, frag_b(kr + 32, hi), z);
      s[nc] = z;
    }

    // online softmax (rows live across 16-lane half-waves, 8 rows per lane)
    float alpha[8];
#pragma unroll
    for (int j = 0; j < 8; ++j) {
      float mx = fmaxf(s[0][j], s[1][j]);
      for (int m = 8; m >= 1; m >>= 1) mx = fmaxf(mx, __shfl_xor(mx, m, 32));
      const float nm = fmaxf(mrow[j], mx);
      const float a = __expf(mrow[j] - nm);
      const float p0 = __expf(s[0][j] - nm);
      const float p1 = __expf(s[1][j] - nm);
      s[0][j] = p0;
      s[1][j] = p1;
      float rs = p0 + p1;
      for (int m = 8; m >= 1; m >>= 1) rs += __shfl_xor(rs, m, 32);
      lrow[j] = lrow[j] * a + rs;
      mrow[j] = nm;
      alpha[j] = a;
    }
#pragma unroll
    for (int f = 0; f < 4; ++f)
#pragma unroll
      for (int j = 0; j < 8; ++j) oacc[f][j] *= alpha[j];

    // restage P (C-fragment layout) into LDS row-major [16][32]
    bf16_t* P = Pl[wid];
#pragma unroll
    for (int j = 0; j < 8; ++j) {
      const int m = j + 8 * hi;
      P[m * 32 + lo] = (bf16_t)s[0][j];
      P[m * 32 + 16 + lo] = (bf16_t)s[1][j];
    }
    __syncthreads();

    // O(16x64) += P(16x32) * V_tile(32x64)
    const v16bf pa = frag_a(P + lo * 32, hi);
#pragma unroll
    for (int f = 0; f < 4; ++f) {
      v16bf vb;
      const int d = f * 16 + lo;
#pragma unroll
      for (int h = 0; h < 16; ++h) vb[h] = Vt[buf][(h < 8 ? hi * 8 + h : 8 + hi * 8 + h) * 64 + d];
      oacc[f] = wmma_bf16(pa, vb, oacc[f]);
    }
  }

  const int b_ = bh / HEADS, h_ = bh % HEADS;
#pragma unroll
  for (int f = 0; f < 4; ++f) {
#pragma unroll
    for (int j = 0; j < 8; ++j) {
      const int m = j + 8 * hi;
      const size_t row = (size_t)b_ * SEQ + row0 + m;
      const int col = h_ * DH + f * 16 + lo;
      o[row * DIM + col] = (bf16_t)(oacc[f][j] / lrow[j]);
    }
  }
}

// ---------------------------------------------------------------------------
extern "C" void kernel_launch(void* const* d_in, const int* in_sizes, int n_in,
                              void* d_out, int out_size, void* d_ws,
                              size_t ws_size, hipStream_t stream) {
  (void)in_sizes; (void)n_in; (void)out_size; (void)ws_size;
  const float* x      = (const float*)d_in[0];
  const float* ln1w   = (const float*)d_in[1];
  const float* ln1b   = (const float*)d_in[2];
  const float* qkvw   = (const float*)d_in[3];
  const float* qkvb   = (const float*)d_in[4];
  const float* projw  = (const float*)d_in[5];
  const float* projb  = (const float*)d_in[6];
  const float* gamma1 = (const float*)d_in[7];
  const float* ln2w   = (const float*)d_in[8];
  const float* ln2b   = (const float*)d_in[9];
  const float* fc1w   = (const float*)d_in[10];
  const float* fc1b   = (const float*)d_in[11];
  const float* fc2w   = (const float*)d_in[12];
  const float* fc2b   = (const float*)d_in[13];
  const float* gamma2 = (const float*)d_in[14];
  float* out = (float*)d_out;

  char* p = (char*)d_ws;
  auto take = [&](size_t bytes) -> char* {
    char* r = p;
    p += (bytes + 255) & ~(size_t)255;
    return r;
  };
  bf16_t* qkv_wt  = (bf16_t*)take((size_t)2304 * 768 * 2);
  bf16_t* proj_wt = (bf16_t*)take((size_t)768 * 768 * 2);
  bf16_t* fc1_wt  = (bf16_t*)take((size_t)3072 * 768 * 2);
  bf16_t* fc2_wt  = (bf16_t*)take((size_t)768 * 3072 * 2);
  bf16_t* h1b = (bf16_t*)take((size_t)ROWS * DIM * 2);
  bf16_t* qb  = (bf16_t*)take((size_t)ROWS * DIM * 2);
  bf16_t* kb  = (bf16_t*)take((size_t)ROWS * DIM * 2);
  bf16_t* vb  = (bf16_t*)take((size_t)ROWS * DIM * 2);
  bf16_t* ob  = (bf16_t*)take((size_t)ROWS * DIM * 2);
  float*  x1  = (float*)take((size_t)ROWS * DIM * 4);
  bf16_t* h2b = (bf16_t*)take((size_t)ROWS * DIM * 2);
  bf16_t* hg  = (bf16_t*)take((size_t)ROWS * HIDDEN * 2);

  convert_wt<<<(768 * 2304) / 256, 256, 0, stream>>>(qkvw, qkv_wt, 768, 2304);
  convert_wt<<<(768 * 768) / 256, 256, 0, stream>>>(projw, proj_wt, 768, 768);
  convert_wt<<<(768 * 3072) / 256, 256, 0, stream>>>(fc1w, fc1_wt, 768, 3072);
  convert_wt<<<(3072 * 768) / 256, 256, 0, stream>>>(fc2w, fc2_wt, 3072, 768);

  ln_kernel<<<ROWS / 8, 256, 0, stream>>>(x, ln1w, ln1b, h1b);

  gemm_bf16<0><<<dim3(64, 18), 256, 0, stream>>>(h1b, qkv_wt, qkvb, ROWS, 2304,
                                                 768, qb, kb, vb, nullptr,
                                                 nullptr, nullptr);

  attn_kernel<<<dim3(96, 8), 256, 0, stream>>>(qb, kb, vb, ob);

  gemm_bf16<1><<<dim3(64, 6), 256, 0, stream>>>(ob, proj_wt, projb, ROWS, 768,
                                                768, nullptr, nullptr, nullptr,
                                                x1, x, gamma1);

  ln_kernel<<<ROWS / 8, 256, 0, stream>>>(x1, ln2w, ln2b, h2b);

  gemm_bf16<2><<<dim3(64, 24), 256, 0, stream>>>(h2b, fc1_wt, fc1b, ROWS,
                                                 HIDDEN, 768, hg, nullptr,
                                                 nullptr, nullptr, nullptr,
                                                 nullptr);

  gemm_bf16<1><<<dim3(64, 6), 256, 0, stream>>>(hg, fc2_wt, fc2b, ROWS, 768,
                                                HIDDEN, nullptr, nullptr,
                                                nullptr, out, x1, gamma2);
}
